// ARU_64106681860394
// MI455X (gfx1250) — matrix operand order repaired and checked by reading, not compiled, not run on hardware
//
#include <hip/hip_runtime.h>
#include <hip/hip_bf16.h>

#define HID 1024
#define NB  32
#define NS  1024
#define G3  3072
#define SCAN_WGS 32
#define SCAN_THREADS 128
#define SCAN_LDS_BYTES (2*3*64*16*32)   // 196608 B = 192 KiB per WG

typedef __bf16 bf16_t;
typedef __attribute__((ext_vector_type(8)))  __bf16 v8bf;
typedef __attribute__((ext_vector_type(16))) __bf16 v16bf;
typedef __attribute__((ext_vector_type(8)))  float  v8f;

__device__ __forceinline__ v16bf ld_frag(const bf16_t* p0, const bf16_t* p1) {
  v16bf r;
  ((v8bf*)&r)[0] = *(const v8bf*)p0;
  ((v8bf*)&r)[1] = *(const v8bf*)p1;
  return r;
}

// sigmoid: v_exp_f32 + v_rcp_f32 (no branchy libm divide)
__device__ __forceinline__ float sigmoidf_(float x) {
  return __builtin_amdgcn_rcpf(1.0f + __expf(-x));
}

// tanh: native CDNA5 V_TANH_F32 when the builtin exists, else branchless exp form
__device__ __forceinline__ float fast_tanhf(float x) {
#if __has_builtin(__builtin_amdgcn_tanhf)
  return __builtin_amdgcn_tanhf(x);
#else
  float ax = __builtin_fabsf(x);
  float t  = __expf(-2.0f * ax);                       // in (0,1]
  float r  = 1.0f - 2.0f * t * __builtin_amdgcn_rcpf(1.0f + t);
  return __builtin_copysignf(r, x);
#endif
}

// ---------------------------------------------------------------------------
// Pack W (K x N, f32 row-major) into WMMA-B-native bf16 chunks:
// chunk (n, g) = 16 consecutive-K bf16 values W[g*16 .. g*16+15][n],
// stored at out[ (n*(K/16) + g) * 16 ].  Lane L of a B fragment for
// (kt, nt) then reads its 32B chunk with two contiguous b128 loads.
// ---------------------------------------------------------------------------
__global__ void pack_b_kernel(const float* __restrict__ W, bf16_t* __restrict__ out,
                              int K, int N) {
  int t = blockIdx.x * blockDim.x + threadIdx.x;
  int kg = K >> 4;
  int total = N * kg;
  if (t >= total) return;
  int n = t % N;
  int g = t / N;
  bf16_t* dst = out + ((size_t)n * kg + g) * 16;
  const float* src = W + (size_t)(g * 16) * N + n;
#pragma unroll
  for (int i = 0; i < 16; ++i) dst[i] = (bf16_t)src[(size_t)i * N];
}

// ---------------------------------------------------------------------------
// Embedding gather + f32 -> bf16 (row-major A matrix for phase-1 GEMMs)
// ---------------------------------------------------------------------------
__global__ void gather_kernel(const int* __restrict__ x, const float* __restrict__ embed,
                              bf16_t* __restrict__ e) {
  int t = blockIdx.x * blockDim.x + threadIdx.x;           // one thread = 8 elems
  int row = t >> 7;                                        // 32768 rows
  int h0  = (t & 127) << 3;
  int tok = x[row];
  const float* src = embed + (size_t)tok * HID + h0;
  bf16_t* dst = e + (size_t)row * HID + h0;
#pragma unroll
  for (int i = 0; i < 8; ++i) dst[i] = (bf16_t)src[i];
}

// ---------------------------------------------------------------------------
// Phase-1 GEMM: C[M,N] = act(A[M,K] @ B[K,N] + bias), A bf16 row-major,
// B packed (see pack_b_kernel), C f32. act: 0 = identity, 1 = tanh.
// WG = 256 thr (8 waves), WG tile 128x64, wave tile 32x32 (2x2 WMMA tiles).
// ---------------------------------------------------------------------------
__global__ void gemm_kernel(const bf16_t* __restrict__ A, const bf16_t* __restrict__ Bp,
                            const float* __restrict__ bias, float* __restrict__ C,
                            int M, int N, int K, int act) {
  const int tid  = threadIdx.x;
  const int wave = tid >> 5;
  const int lane = tid & 31;
  const int half = lane >> 4;          // 0: lanes 0-15, 1: lanes 16-31
  const int l16  = lane & 15;
  const int mbase = blockIdx.y * 128 + (wave & 3) * 32;
  const int nbase = blockIdx.x * 64  + (wave >> 2) * 32;
  const int kg16  = K >> 4;
  (void)M;

  v8f acc[2][2];
#pragma unroll
  for (int i = 0; i < 2; ++i)
#pragma unroll
    for (int j = 0; j < 2; ++j)
      acc[i][j] = v8f{0.f,0.f,0.f,0.f,0.f,0.f,0.f,0.f};

  for (int kt = 0; kt < (K >> 5); ++kt) {
    const int kb = kt * 32 + half * 16;
    v16bf a[2], b[2];
#pragma unroll
    for (int ms = 0; ms < 2; ++ms) {
      const bf16_t* ap = A + (size_t)(mbase + ms * 16 + l16) * K + kb;
      __builtin_prefetch(ap + 32, 0, 1);                  // next K-slab -> global_prefetch
      a[ms] = ld_frag(ap, ap + 8);
    }
#pragma unroll
    for (int ns = 0; ns < 2; ++ns) {
      const bf16_t* bp = Bp + ((size_t)(nbase + ns * 16 + l16) * kg16 + kt * 2 + half) * 16;
      b[ns] = ld_frag(bp, bp + 8);
    }
#pragma unroll
    for (int ms = 0; ms < 2; ++ms)
#pragma unroll
      for (int ns = 0; ns < 2; ++ns)
        acc[ms][ns] = __builtin_amdgcn_wmma_f32_16x16x32_bf16(
            false, a[ms], false, b[ns], (short)0, acc[ms][ns], false, false);
  }

#pragma unroll
  for (int ms = 0; ms < 2; ++ms)
#pragma unroll
    for (int ns = 0; ns < 2; ++ns)
#pragma unroll
      for (int i = 0; i < 8; ++i) {
        int row = mbase + ms * 16 + i + half * 8;
        int col = nbase + ns * 16 + l16;
        float v = acc[ms][ns][i] + bias[col];
        if (act) v = fast_tanhf(v);
        C[(size_t)row * N + col] = v;
      }
}

// ---------------------------------------------------------------------------
// Software grid barrier (32 WGs, persistent kernel; trivially co-resident)
// ---------------------------------------------------------------------------
__device__ __forceinline__ void grid_sync(unsigned* cnt, unsigned target) {
  __threadfence();
  __syncthreads();
  if (threadIdx.x == 0) {
    __hip_atomic_fetch_add(cnt, 1u, __ATOMIC_RELEASE, __HIP_MEMORY_SCOPE_AGENT);
    while (__hip_atomic_load(cnt, __ATOMIC_ACQUIRE, __HIP_MEMORY_SCOPE_AGENT) < target)
      __builtin_amdgcn_s_sleep(1);
  }
  __syncthreads();
  __threadfence();
}

// ---------------------------------------------------------------------------
// Recurrent scan.  Grid = 32 WGs x 128 thr (4 waves).  Each WG owns two
// 16-column triples of (rho, pi, alpha); its 192 KiB slice of packed Wgh
// is staged into LDS ONCE (async L2->LDS path, ASYNCcnt-tracked) and reused
// for all 1024 steps.  Wave w handles col-triple (w>>1) and M-tile (w&1):
// 3 accumulators (rho/pi/alpha).  h (f32) lives in d_out; bf16 h is
// double-buffered -> 1 grid barrier per step.
// ---------------------------------------------------------------------------
__global__ void scan_kernel(const bf16_t* __restrict__ Wp,   // packed Wgh (K=1024, N=3072)
                            const float* __restrict__ gx,    // gate_x (B,S,3H) f32
                            const float* __restrict__ cand,  // cand   (B,S,H)  f32
                            float* __restrict__ h,           // (B,H) f32 == d_out
                            bf16_t* __restrict__ hb0,
                            bf16_t* __restrict__ hb1,
                            unsigned* __restrict__ cnt) {
  extern __shared__ bf16_t lds[];
  const int wg   = blockIdx.x;
  const int tid  = threadIdx.x;
  const int wave = tid >> 5;
  const int lane = tid & 31;
  const int half = lane >> 4;
  const int l16  = lane & 15;

  // ---- stage this WG's Wgh slice into LDS (6144 chunks of 16 B) ----
  // Uses CDNA5 async global->LDS DMA: no VGPR round-trip, ASYNCcnt-tracked.
  for (int i = tid; i < 6144; i += SCAN_THREADS) {
    int c   = i & 15;
    int g   = (i >> 4) & 63;
    int rem = i >> 10;            // 0..5 = tp*3 + bl
    int bl  = rem % 3;
    int tp  = rem / 3;
    int n   = bl * HID + (wg * 2 + tp) * 16 + c;
    const bf16_t* gsrc = Wp + ((size_t)n * 64 + g) * 16;
    unsigned ldsoff = (unsigned)(size_t)(void*)(lds + (size_t)i * 16);  // LDS byte offset
    asm volatile("global_load_async_to_lds_b128 %0, %1, off"
                 :: "v"(ldsoff), "v"(gsrc) : "memory");
  }
#if __has_builtin(__builtin_amdgcn_s_wait_asynccnt)
  __builtin_amdgcn_s_wait_asynccnt(0);
#else
  asm volatile("s_wait_asynccnt 0" ::: "memory");
#endif
  __syncthreads();

  const int tp = wave >> 1;              // which col-triple of this WG
  const int mt = wave & 1;               // which 16-row half of B(=32) rows
  const int n0 = (wg * 2 + tp) * 16;     // hidden-column base [0,1024)
  const bf16_t* ldsB = lds + (size_t)tp * 3 * 64 * 16 * 16;

  for (int t = 0; t < NS; ++t) {
    const bf16_t* hr = (t & 1) ? hb1 : hb0;
    bf16_t*       hw = (t & 1) ? hb0 : hb1;

    v8f acc[3];
#pragma unroll
    for (int b = 0; b < 3; ++b) acc[b] = v8f{0.f,0.f,0.f,0.f,0.f,0.f,0.f,0.f};

    const int rowA = mt * 16 + l16;
    const bf16_t* abase = hr + (size_t)rowA * HID + half * 16;
    for (int kt = 0; kt < 32; ++kt) {
      const bf16_t* ap = abase + kt * 32;
      v16bf a = ld_frag(ap, ap + 8);
#pragma unroll
      for (int bl = 0; bl < 3; ++bl) {
        const bf16_t* bp = ldsB + ((size_t)(bl * 64 + kt * 2 + half) * 16 + l16) * 16;
        v16bf b = ld_frag(bp, bp + 8);
        acc[bl] = __builtin_amdgcn_wmma_f32_16x16x32_bf16(
            false, a, false, b, (short)0, acc[bl], false, false);
      }
    }

    // ---- gate nonlinearity + state update (elements owned exclusively) ----
#pragma unroll
    for (int i = 0; i < 8; ++i) {
      int row = mt * 16 + i + half * 8;      // batch index b
      int col = n0 + l16;                    // hidden index
      size_t gbase = ((size_t)row * NS + t) * G3 + col;
      float r  = sigmoidf_(acc[0][i] + gx[gbase]);
      float p  = sigmoidf_(acc[1][i] + gx[gbase + HID]);
      float al = sigmoidf_(acc[2][i] + gx[gbase + 2 * HID]);
      float v  = cand[((size_t)row * NS + t) * HID + col];
      float ho = h[(size_t)row * HID + col];
      float hn = r * (p * ho + al * v);
      h[(size_t)row * HID + col]  = hn;
      hw[(size_t)row * HID + col] = (bf16_t)hn;
    }

    grid_sync(cnt, (unsigned)(SCAN_WGS * (t + 1)));
  }
}

// ---------------------------------------------------------------------------
extern "C" void kernel_launch(void* const* d_in, const int* in_sizes, int n_in,
                              void* d_out, int out_size, void* d_ws, size_t ws_size,
                              hipStream_t stream) {
  (void)in_sizes; (void)n_in; (void)out_size; (void)ws_size;
  const int*   x     = (const int*)  d_in[0];
  const float* embed = (const float*)d_in[1];
  const float* Wc    = (const float*)d_in[2];
  const float* bc    = (const float*)d_in[3];
  const float* Wgx   = (const float*)d_in[4];
  const float* bgx   = (const float*)d_in[5];
  const float* Wgh   = (const float*)d_in[6];
  float* h_out = (float*)d_out;

  // ---- workspace layout ----
  char* ws = (char*)d_ws;
  size_t off = 0;
  auto take = [&](size_t bytes) { char* p = ws + off; off += (bytes + 255) & ~(size_t)255; return p; };
  bf16_t* e_bf   = (bf16_t*)take((size_t)NB * NS * HID * 2);          // 64 MiB
  bf16_t* Wc_p   = (bf16_t*)take((size_t)HID * HID * 2);              // 2 MiB
  bf16_t* Wgx_p  = (bf16_t*)take((size_t)HID * G3 * 2);               // 6 MiB
  bf16_t* Wgh_p  = (bf16_t*)take((size_t)HID * G3 * 2);               // 6 MiB
  float*  gate_x = (float*) take((size_t)NB * NS * G3 * 4);           // 384 MiB
  float*  candv  = (float*) take((size_t)NB * NS * HID * 4);          // 128 MiB
  bf16_t* hb0    = (bf16_t*)take((size_t)NB * HID * 2);
  bf16_t* hb1    = (bf16_t*)take((size_t)NB * HID * 2);
  unsigned* cnt  = (unsigned*)take(256);

  (void)hipMemsetAsync(hb0, 0, (size_t)NB * HID * 2, stream);
  (void)hipMemsetAsync(hb1, 0, (size_t)NB * HID * 2, stream);
  (void)hipMemsetAsync(cnt, 0, 256, stream);
  (void)hipMemsetAsync(h_out, 0, (size_t)NB * HID * 4, stream);

  // ---- pack weights to bf16 WMMA-B layout ----
  {
    int tot = HID * (HID >> 4);
    pack_b_kernel<<<(tot + 255) / 256, 256, 0, stream>>>(Wc, Wc_p, HID, HID);
    tot = G3 * (HID >> 4);
    pack_b_kernel<<<(tot + 255) / 256, 256, 0, stream>>>(Wgx, Wgx_p, HID, G3);
    pack_b_kernel<<<(tot + 255) / 256, 256, 0, stream>>>(Wgh, Wgh_p, HID, G3);
  }

  // ---- embedding gather -> bf16 ----
  {
    int tot = NB * NS * (HID / 8);
    gather_kernel<<<tot / 256, 256, 0, stream>>>(x, embed, e_bf);
  }

  // ---- phase-1 GEMMs ----
  {
    dim3 g1(G3 / 64, (NB * NS) / 128);
    gemm_kernel<<<g1, 256, 0, stream>>>(e_bf, Wgx_p, bgx, gate_x, NB * NS, G3, HID, 0);
    dim3 g2(HID / 64, (NB * NS) / 128);
    gemm_kernel<<<g2, 256, 0, stream>>>(e_bf, Wc_p, bc, candv, NB * NS, HID, HID, 1);
  }

  // ---- recurrent scan (persistent, LDS-resident Wgh) ----
  (void)hipFuncSetAttribute(reinterpret_cast<const void*>(scan_kernel),
                            hipFuncAttributeMaxDynamicSharedMemorySize, SCAN_LDS_BYTES);
  scan_kernel<<<SCAN_WGS, SCAN_THREADS, SCAN_LDS_BYTES, stream>>>(
      Wgh_p, gate_x, candv, h_out, hb0, hb1, cnt);
}
